// TortoiseDiscreteTokenInjector_49039936586452
// MI455X (gfx1250) — matrix operand order repaired and checked by reading, not compiled, not run on hardware
//
#include <hip/hip_runtime.h>

// ---------------------------------------------------------------------------
// Types for CDNA5 WMMA (gfx1250, wave32)
// ---------------------------------------------------------------------------
typedef __attribute__((ext_vector_type(16))) __bf16 v16bf;
typedef __attribute__((ext_vector_type(8)))  float  v8f;

union FragU { v16bf v; uint4 q[2]; };

__device__ __forceinline__ unsigned short f2bf(float f) {
  unsigned u = __float_as_uint(f);
  unsigned r = ((u >> 16) & 1u) + 0x7FFFu;   // round-to-nearest-even
  return (unsigned short)((u + r) >> 16);
}
__device__ __forceinline__ float bf2f(unsigned short h) {
  return __uint_as_float(((unsigned)h) << 16);
}

// CDNA5 async global->LDS DMA (no data VGPRs, tracked by ASYNCcnt).
// LDS operand = low 32 bits of the flat shared-aperture address.
__device__ __forceinline__ void async_load_b128(const void* g, void* l) {
  unsigned long long ga = (unsigned long long)g;
  unsigned la = (unsigned)(unsigned long long)l;
  asm volatile("global_load_async_to_lds_b128 %0, %1, off"
               :: "v"(la), "v"(ga) : "memory");
}
__device__ __forceinline__ void wait_async0() {
  asm volatile("s_wait_asynccnt 0" ::: "memory");
}

// ---------------------------------------------------------------------------
// Small helper kernels
// ---------------------------------------------------------------------------
__global__ void zero_pads_kernel(unsigned short* buf, int Tp2, int C) {
  int i = blockIdx.x * blockDim.x + threadIdx.x;
  int total = 16 * 2 * C;
  if (i >= total) return;
  int b = i / (2 * C);
  int rem = i - b * (2 * C);
  int which = rem / C;
  int c = rem - which * C;
  int row = which ? (Tp2 - 1) : 0;
  buf[((size_t)b * Tp2 + row) * C + c] = 0;
}

__global__ void init_keys_kernel(unsigned long long* keys) {
  int i = blockIdx.x * blockDim.x + threadIdx.x;
  if (i < 8192) keys[i] = ~0ull;
}

// w[O][Ci][K] (f32) -> wt[O][K*Ci] (bf16), K-major so it matches the
// contiguous [t-1,t,t+1] activation patch layout.
__global__ void wt_convert_kernel(const float* __restrict__ w,
                                  unsigned short* __restrict__ wt,
                                  int O, int Ci, int K) {
  int idx = blockIdx.x * blockDim.x + threadIdx.x;
  int total = O * Ci * K;
  if (idx >= total) return;
  int o = idx / (Ci * K);
  int rem = idx - o * (Ci * K);
  int ci = rem / K;
  int k = rem - ci * K;
  wt[(size_t)o * (Ci * K) + k * Ci + ci] = f2bf(w[idx]);
}

// codebook [512][8192] f32 -> cbT [8192][512] bf16, plus per-code squared norm
__global__ void cb_convert_kernel(const float* __restrict__ cb,
                                  unsigned short* __restrict__ cbT,
                                  float* __restrict__ cnorm) {
  int j = blockIdx.x * blockDim.x + threadIdx.x;
  if (j >= 8192) return;
  float s = 0.f;
  for (int d = 0; d < 512; ++d) {
    float v = cb[(size_t)d * 8192 + j];
    s += v * v;
    cbT[(size_t)j * 512 + d] = f2bf(v);
  }
  cnorm[j] = s;
}

// conv1: x[16][80][2048] f32, w1[512][80][3], stride 2 pad 1, relu ->
// h1 padded [16][1026][512] bf16.  <1% of FLOPs -> simple VALU kernel.
__global__ void __launch_bounds__(256) conv1_kernel(
    const float* __restrict__ x, const float* __restrict__ w1,
    const float* __restrict__ b1, unsigned short* __restrict__ h1) {
  __shared__ float patch[240];
  const int t = blockIdx.x;   // 0..1023
  const int b = blockIdx.y;   // 0..15
  const int tid = threadIdx.x;
  if (tid < 240) {
    int ci = tid / 3, k = tid - ci * 3;
    int pos = 2 * t - 1 + k;
    patch[tid] = (pos >= 0 && pos < 2048)
                     ? x[((size_t)b * 80 + ci) * 2048 + pos] : 0.f;
  }
  __syncthreads();
  for (int o = tid; o < 512; o += 256) {
    const float* wr = w1 + (size_t)o * 240;
    float acc = b1[o];
#pragma unroll 4
    for (int j = 0; j < 240; ++j) acc += patch[j] * wr[j];
    acc = acc > 0.f ? acc : 0.f;
    h1[((size_t)b * 1026 + t + 1) * 512 + o] = f2bf(acc);
  }
}

// ---------------------------------------------------------------------------
// Generic conv-as-GEMM, bf16 WMMA, f32 accumulate.
// A: padded activations [B][Tin+2][Cin] bf16; Bt: weights [Cout][Ktot] bf16.
// Block tile 64(M) x 256(N) x 32(K), double-buffered LDS filled by
// GLOBAL_LOAD_ASYNC_TO_LDS_B128 (ASYNCcnt), 1 barrier per K-step.
// 8 waves in 2(M) x 4(N); each wave: 32x64 C tile = 8 accumulators,
// 8 WMMA per K-step (each A-frag reused 4x).
// EPI: 0 = bias+relu, 1 = bias, 2 = bias + residual add (Out += ...)
// ---------------------------------------------------------------------------
template <int EPI>
__global__ void __launch_bounds__(256) gemm_conv_kernel(
    const unsigned short* __restrict__ A,
    const unsigned short* __restrict__ Bt,
    const float* __restrict__ bias,
    unsigned short* __restrict__ Out,
    int Tin, int Tout, int Cin, int Cout, int Ktot,
    int convStride, int rowAdd) {
  __shared__ __align__(16) unsigned short As[2][64 * 40];
  __shared__ __align__(16) unsigned short Bs[2][256 * 40];

  const int tid = threadIdx.x;
  const int lane = tid & 31;
  const int wm = (tid >> 5) >> 2;   // 0..1
  const int wn = (tid >> 5) & 3;    // 0..3
  const int blockN = blockIdx.x * 256;
  const int blockM = blockIdx.y * 64;

  // A tile loader: 64 rows x 32 K, one b128 per thread
  const int arow = tid >> 2;
  const int achunk = (tid & 3) << 3;
  const int am = blockM + arow;
  const int ab = am / Tout;
  const int at = am - ab * Tout;
  const unsigned short* aPtr =
      A + (size_t)(ab * (Tin + 2) + at * convStride + rowAdd) * Cin + achunk;

  // B tile loader: 256 rows x 32 K, one full row (4 x b128) per thread
  const unsigned short* bPtr = Bt + (size_t)(blockN + tid) * Ktot;

  const int l15 = lane & 15;
  const int hi = lane >> 4;

  v8f acc[2][4];
  {
    v8f z = {};
#pragma unroll
    for (int i = 0; i < 2; ++i)
#pragma unroll
      for (int j = 0; j < 4; ++j) acc[i][j] = z;
  }

  const int nk = Ktot >> 5;

  // prologue: async-fill K-tile 0 into LDS buffer 0
  async_load_b128(aPtr, &As[0][arow * 40 + achunk]);
  async_load_b128(bPtr + 0,  &Bs[0][tid * 40 + 0]);
  async_load_b128(bPtr + 8,  &Bs[0][tid * 40 + 8]);
  async_load_b128(bPtr + 16, &Bs[0][tid * 40 + 16]);
  async_load_b128(bPtr + 24, &Bs[0][tid * 40 + 24]);

  for (int kk = 0; kk < nk; ++kk) {
    const int buf = kk & 1;
    const int nxt = buf ^ 1;
    const bool more = (kk + 1) < nk;

    wait_async0();        // my writes of tile kk have landed in LDS
    __syncthreads();      // everyone's writes landed; buf^1 readers drained

    if (more) {           // DMA next tile while this one is consumed
      const int k0 = (kk + 1) << 5;
      async_load_b128(aPtr + k0, &As[nxt][arow * 40 + achunk]);
      async_load_b128(bPtr + k0 + 0,  &Bs[nxt][tid * 40 + 0]);
      async_load_b128(bPtr + k0 + 8,  &Bs[nxt][tid * 40 + 8]);
      async_load_b128(bPtr + k0 + 16, &Bs[nxt][tid * 40 + 16]);
      async_load_b128(bPtr + k0 + 24, &Bs[nxt][tid * 40 + 24]);
    }

    // A fragments: lane L -> row L&15, K chunks {koff..koff+7, 16+koff..}
    FragU fa0, fa1;
    const unsigned short* pa0 = &As[buf][(wm * 32 + l15) * 40 + hi * 8];
    fa0.q[0] = *reinterpret_cast<const uint4*>(pa0);
    fa0.q[1] = *reinterpret_cast<const uint4*>(pa0 + 16);
    const unsigned short* pa1 = pa0 + 16 * 40;
    fa1.q[0] = *reinterpret_cast<const uint4*>(pa1);
    fa1.q[1] = *reinterpret_cast<const uint4*>(pa1 + 16);

#pragma unroll
    for (int j = 0; j < 4; ++j) {
      // B fragment: lane L -> col L&15, K chunks {koff..koff+7, 16+koff..}
      FragU fb;
      const unsigned short* pb =
          &Bs[buf][(wn * 64 + j * 16 + l15) * 40 + hi * 8];
      fb.q[0] = *reinterpret_cast<const uint4*>(pb);
      fb.q[1] = *reinterpret_cast<const uint4*>(pb + 16);
      acc[0][j] = __builtin_amdgcn_wmma_f32_16x16x32_bf16(
          false, fa0.v, false, fb.v, (short)0, acc[0][j], false, false);
      acc[1][j] = __builtin_amdgcn_wmma_f32_16x16x32_bf16(
          false, fa1.v, false, fb.v, (short)0, acc[1][j], false, false);
    }
  }

  const int hi8 = hi << 3;
#pragma unroll
  for (int mi = 0; mi < 2; ++mi)
#pragma unroll
    for (int nj = 0; nj < 4; ++nj) {
      v8f a = acc[mi][nj];
#pragma unroll
      for (int r = 0; r < 8; ++r) {
        const int m = blockM + wm * 32 + mi * 16 + r + hi8;
        const int n = blockN + wn * 64 + nj * 16 + l15;
        const int b = m / Tout;
        const int t = m - b * Tout;
        const size_t oo = (size_t)(b * (Tout + 2) + t + 1) * Cout + n;
        float v = a[r] + bias[n];
        if (EPI == 0) v = v > 0.f ? v : 0.f;
        if (EPI == 2) v += bf2f(Out[oo]);
        Out[oo] = f2bf(v);
      }
    }
}

// ---------------------------------------------------------------------------
// Codebook GEMM + fused argmin:  dist_j = ||c_j||^2 - 2 f.c_j  (||f||^2 const)
// key = (orderable(dist) << 32) | code_index ; global u64 atomicMin per row.
// Same 64x256x32 async double-buffered WMMA core.
// ---------------------------------------------------------------------------
__global__ void __launch_bounds__(256) vq_argmin_kernel(
    const unsigned short* __restrict__ lat,   // [16][514][512] bf16 padded
    const unsigned short* __restrict__ cbT,   // [8192][512] bf16
    const float* __restrict__ cnorm,          // [8192]
    unsigned long long* __restrict__ keys) {  // [8192]
  __shared__ __align__(16) unsigned short As[2][64 * 40];
  __shared__ __align__(16) unsigned short Bs[2][256 * 40];
  __shared__ unsigned long long rowmin[64];

  const int tid = threadIdx.x;
  const int lane = tid & 31;
  const int wm = (tid >> 5) >> 2;
  const int wn = (tid >> 5) & 3;
  const int blockN = blockIdx.x * 256;
  const int blockM = blockIdx.y * 64;

  if (tid < 64) rowmin[tid] = ~0ull;

  const int arow = tid >> 2;
  const int achunk = (tid & 3) << 3;
  const int am = blockM + arow;
  const int ab = am >> 9;
  const int at = am & 511;
  const unsigned short* aPtr = lat + (size_t)(ab * 514 + at + 1) * 512 + achunk;
  const unsigned short* bPtr = cbT + (size_t)(blockN + tid) * 512;

  const int l15 = lane & 15;
  const int hi = lane >> 4;

  v8f acc[2][4];
  {
    v8f z = {};
#pragma unroll
    for (int i = 0; i < 2; ++i)
#pragma unroll
      for (int j = 0; j < 4; ++j) acc[i][j] = z;
  }

  async_load_b128(aPtr, &As[0][arow * 40 + achunk]);
  async_load_b128(bPtr + 0,  &Bs[0][tid * 40 + 0]);
  async_load_b128(bPtr + 8,  &Bs[0][tid * 40 + 8]);
  async_load_b128(bPtr + 16, &Bs[0][tid * 40 + 16]);
  async_load_b128(bPtr + 24, &Bs[0][tid * 40 + 24]);

  for (int kk = 0; kk < 16; ++kk) {   // 512 / 32
    const int buf = kk & 1;
    const int nxt = buf ^ 1;
    const bool more = (kk + 1) < 16;

    wait_async0();
    __syncthreads();

    if (more) {
      const int k0 = (kk + 1) << 5;
      async_load_b128(aPtr + k0, &As[nxt][arow * 40 + achunk]);
      async_load_b128(bPtr + k0 + 0,  &Bs[nxt][tid * 40 + 0]);
      async_load_b128(bPtr + k0 + 8,  &Bs[nxt][tid * 40 + 8]);
      async_load_b128(bPtr + k0 + 16, &Bs[nxt][tid * 40 + 16]);
      async_load_b128(bPtr + k0 + 24, &Bs[nxt][tid * 40 + 24]);
    }

    FragU fa0, fa1;
    const unsigned short* pa0 = &As[buf][(wm * 32 + l15) * 40 + hi * 8];
    fa0.q[0] = *reinterpret_cast<const uint4*>(pa0);
    fa0.q[1] = *reinterpret_cast<const uint4*>(pa0 + 16);
    const unsigned short* pa1 = pa0 + 16 * 40;
    fa1.q[0] = *reinterpret_cast<const uint4*>(pa1);
    fa1.q[1] = *reinterpret_cast<const uint4*>(pa1 + 16);

#pragma unroll
    for (int j = 0; j < 4; ++j) {
      FragU fb;
      const unsigned short* pb =
          &Bs[buf][(wn * 64 + j * 16 + l15) * 40 + hi * 8];
      fb.q[0] = *reinterpret_cast<const uint4*>(pb);
      fb.q[1] = *reinterpret_cast<const uint4*>(pb + 16);
      acc[0][j] = __builtin_amdgcn_wmma_f32_16x16x32_bf16(
          false, fa0.v, false, fb.v, (short)0, acc[0][j], false, false);
      acc[1][j] = __builtin_amdgcn_wmma_f32_16x16x32_bf16(
          false, fa1.v, false, fb.v, (short)0, acc[1][j], false, false);
    }
  }

  const int hi8 = hi << 3;
#pragma unroll
  for (int mi = 0; mi < 2; ++mi)
#pragma unroll
    for (int nj = 0; nj < 4; ++nj) {
      v8f a = acc[mi][nj];
#pragma unroll
      for (int r = 0; r < 8; ++r) {
        const int row = wm * 32 + mi * 16 + r + hi8;
        const int n = blockN + wn * 64 + nj * 16 + l15;
        float dist = cnorm[n] - 2.0f * a[r];
        unsigned u = __float_as_uint(dist);
        u = (u & 0x80000000u) ? ~u : (u | 0x80000000u);
        unsigned long long key =
            ((unsigned long long)u << 32) | (unsigned long long)(unsigned)n;
#pragma unroll
        for (int s = 1; s < 16; s <<= 1) {
          unsigned long long o = __shfl_xor(key, s, 32);
          if (o < key) key = o;
        }
        if (l15 == 0) atomicMin(&rowmin[row], key);
      }
    }
  __syncthreads();
  if (tid < 64) atomicMin(&keys[blockM + tid], rowmin[tid]);
}

__global__ void decode_kernel(const unsigned long long* __restrict__ keys,
                              int* __restrict__ codes) {
  int i = blockIdx.x * blockDim.x + threadIdx.x;
  if (i < 8192) codes[i] = (int)(keys[i] & 0xffffffffull);
}

// ---------------------------------------------------------------------------
// Host-side orchestration
// ---------------------------------------------------------------------------
extern "C" void kernel_launch(void* const* d_in, const int* in_sizes, int n_in,
                              void* d_out, int out_size, void* d_ws,
                              size_t ws_size, hipStream_t stream) {
  (void)in_sizes; (void)n_in; (void)out_size; (void)ws_size;
  const float* x   = (const float*)d_in[0];
  const float* w1  = (const float*)d_in[1];
  const float* b1  = (const float*)d_in[2];
  const float* w2  = (const float*)d_in[3];
  const float* b2  = (const float*)d_in[4];
  const float* rwa = (const float*)d_in[5];
  const float* rba = (const float*)d_in[6];
  const float* rwb = (const float*)d_in[7];
  const float* rbb = (const float*)d_in[8];
  const float* rwc = (const float*)d_in[9];
  const float* rbc = (const float*)d_in[10];
  const float* wf  = (const float*)d_in[11];
  const float* bfv = (const float*)d_in[12];
  const float* cb  = (const float*)d_in[13];
  int* codes = (int*)d_out;

  char* ws = (char*)d_ws;
  size_t off = 0;
  auto alloc = [&](size_t bytes) -> void* {
    void* p = ws + off;
    off = (off + bytes + 255) & ~(size_t)255;
    return p;
  };
  unsigned short* h1   = (unsigned short*)alloc((size_t)16 * 1026 * 512 * 2);
  unsigned short* h2   = (unsigned short*)alloc((size_t)16 * 514 * 1024 * 2);
  unsigned short* r1   = (unsigned short*)alloc((size_t)16 * 514 * 1024 * 2);
  unsigned short* r2   = (unsigned short*)alloc((size_t)16 * 514 * 1024 * 2);
  unsigned short* lat  = (unsigned short*)alloc((size_t)16 * 514 * 512 * 2);
  unsigned short* w2t  = (unsigned short*)alloc((size_t)1024 * 1536 * 2);
  unsigned short* rwat = (unsigned short*)alloc((size_t)3 * 1024 * 3072 * 2);
  unsigned short* rwbt = (unsigned short*)alloc((size_t)3 * 1024 * 3072 * 2);
  unsigned short* rwct = (unsigned short*)alloc((size_t)3 * 1024 * 1024 * 2);
  unsigned short* wft  = (unsigned short*)alloc((size_t)512 * 1024 * 2);
  unsigned short* cbT  = (unsigned short*)alloc((size_t)8192 * 512 * 2);
  float* cnorm = (float*)alloc((size_t)8192 * 4);
  unsigned long long* keys = (unsigned long long*)alloc((size_t)8192 * 8);

  // init pad rows + argmin keys
  zero_pads_kernel<<<(16 * 2 * 512 + 255) / 256, 256, 0, stream>>>(h1, 1026, 512);
  zero_pads_kernel<<<(16 * 2 * 1024 + 255) / 256, 256, 0, stream>>>(h2, 514, 1024);
  zero_pads_kernel<<<(16 * 2 * 1024 + 255) / 256, 256, 0, stream>>>(r1, 514, 1024);
  zero_pads_kernel<<<(16 * 2 * 1024 + 255) / 256, 256, 0, stream>>>(r2, 514, 1024);
  init_keys_kernel<<<32, 256, 0, stream>>>(keys);

  // weight / codebook conversions to bf16 K-major
  wt_convert_kernel<<<(1024 * 512 * 3 + 255) / 256, 256, 0, stream>>>(
      w2, w2t, 1024, 512, 3);
  for (int i = 0; i < 3; ++i) {
    wt_convert_kernel<<<(1024 * 1024 * 3 + 255) / 256, 256, 0, stream>>>(
        rwa + (size_t)i * 1024 * 1024 * 3, rwat + (size_t)i * 1024 * 3072,
        1024, 1024, 3);
    wt_convert_kernel<<<(1024 * 1024 * 3 + 255) / 256, 256, 0, stream>>>(
        rwb + (size_t)i * 1024 * 1024 * 3, rwbt + (size_t)i * 1024 * 3072,
        1024, 1024, 3);
    wt_convert_kernel<<<(1024 * 1024 + 255) / 256, 256, 0, stream>>>(
        rwc + (size_t)i * 1024 * 1024, rwct + (size_t)i * 1024 * 1024,
        1024, 1024, 1);
  }
  wt_convert_kernel<<<(512 * 1024 + 255) / 256, 256, 0, stream>>>(
      wf, wft, 512, 1024, 1);
  cb_convert_kernel<<<8192 / 256, 256, 0, stream>>>(cb, cbT, cnorm);

  // conv1 (80 -> 512, stride 2, relu)
  conv1_kernel<<<dim3(1024, 16), 256, 0, stream>>>(x, w1, b1, h1);

  // conv2 (512 -> 1024, stride 2, relu):  M=8192, N=1024, K=1536
  gemm_conv_kernel<0><<<dim3(4, 128), 256, 0, stream>>>(
      h1, w2t, b2, h2, 1024, 512, 512, 1024, 1536, 2, 0);

  // 3 residual blocks: conv3-relu, conv3-relu, conv1 + residual add
  for (int i = 0; i < 3; ++i) {
    gemm_conv_kernel<0><<<dim3(4, 128), 256, 0, stream>>>(
        h2, rwat + (size_t)i * 1024 * 3072, rba + i * 1024, r1,
        512, 512, 1024, 1024, 3072, 1, 0);
    gemm_conv_kernel<0><<<dim3(4, 128), 256, 0, stream>>>(
        r1, rwbt + (size_t)i * 1024 * 3072, rbb + i * 1024, r2,
        512, 512, 1024, 1024, 3072, 1, 0);
    gemm_conv_kernel<2><<<dim3(4, 128), 256, 0, stream>>>(
        r2, rwct + (size_t)i * 1024 * 1024, rbc + i * 1024, h2,
        512, 512, 1024, 1024, 1024, 1, 1);
  }

  // final 1x1 conv (1024 -> 512):  M=8192, N=512, K=1024
  gemm_conv_kernel<1><<<dim3(2, 128), 256, 0, stream>>>(
      h2, wft, bfv, lat, 512, 512, 1024, 512, 1024, 1, 1);

  // codebook GEMM + fused argmin, then decode
  vq_argmin_kernel<<<dim3(32, 128), 256, 0, stream>>>(lat, cbT, cnorm, keys);
  decode_kernel<<<32, 256, 0, stream>>>(keys, codes);
}
